// MultiHeadAttention_48060684042294
// MI455X (gfx1250) — matrix operand order repaired and checked by reading, not compiled, not run on hardware
//
#include <hip/hip_runtime.h>
#include <hip/hip_bf16.h>

// ---- problem constants (match reference) ----
#define SEQ     2048
#define DM      1024
#define NH      16
#define DKH     64
#define NB      4

typedef __attribute__((ext_vector_type(16))) __bf16 v16bf;
typedef __attribute__((ext_vector_type(8)))  float  v8f;
typedef __attribute__((ext_vector_type(4)))  unsigned int u32x4;
typedef __attribute__((ext_vector_type(8)))  int i32x8;
typedef __attribute__((ext_vector_type(4)))  int i32x4;

union ABu { v16bf v; uint4 q[2]; unsigned short s[16]; };

__device__ __forceinline__ unsigned short bf16rte(float f) {
  unsigned int u = __float_as_uint(f);
  u += 0x7FFFu + ((u >> 16) & 1u);
  return (unsigned short)(u >> 16);
}

__device__ __forceinline__ v8f wmma_bf16(v16bf a, v16bf b, v8f c) {
  return __builtin_amdgcn_wmma_f32_16x16x32_bf16(false, a, false, b, (short)0, c, false, false);
}

// ---- tile loaders -------------------------------------------------------
// A 16x32 bf16 tile: lane = m + 16*h holds row m, K = {8h..8h+7, 16+8h..16+8h+7}.
__device__ __forceinline__ v16bf loadA_bf16(const unsigned short* __restrict__ base,
                                            int ld, int mbase, int k0) {
  int lane = threadIdx.x & 31;
  int m = mbase + (lane & 15);
  int h = lane >> 4;
  const unsigned short* p = base + (size_t)m * ld + k0 + 8 * h;
  ABu a;
  a.q[0] = *(const uint4*)(p);
  a.q[1] = *(const uint4*)(p + 16);
  return a.v;
}

// B 32x16 bf16 tile: lane = n + 16*h holds column n, K = {8h..8h+7, 16+8h..16+8h+7}.
// "base" indexed [column][K] row-major (= W rows / Kp rows / Vt rows / LDS tiles).
__device__ __forceinline__ v16bf loadB_bf16(const unsigned short* base,
                                            int ld, int nbase, int k0) {
  int lane = threadIdx.x & 31;
  int n = nbase + (lane & 15);
  int h = lane >> 4;
  const unsigned short* p = base + (size_t)n * ld + k0 + 8 * h;
  ABu b;
  b.q[0] = *(const uint4*)(p);
  b.q[1] = *(const uint4*)(p + 16);
  return b.v;
}

// 16-lane (half-wave) row reductions for the 16x16 f32 accumulator layout.
__device__ __forceinline__ float rmax16(float v) {
#pragma unroll
  for (int m = 1; m < 16; m <<= 1) v = fmaxf(v, __shfl_xor(v, m, 32));
  return v;
}
__device__ __forceinline__ float rsum16(float v) {
#pragma unroll
  for (int m = 1; m < 16; m <<= 1) v += __shfl_xor(v, m, 32);
  return v;
}

// ---- TDM: 2D tile load Global -> LDS (6-arg clang-23 builtin) -----------
// D# per CDNA5 ISA:
//  group0: [1:0]=count=1 | [63:32]=lds_addr | [120:64]=global_addr | [127:126]=type=2
//  group1: [17:16]=data_size(1=2B) | [79:48]=tensor_dim0 | [111:80]=tensor_dim1
//          [127:112]=tile_dim0 | [143:128]=tile_dim1 | [207:160]=dim0_stride
__device__ __forceinline__ void tdm_load_2d_bf16(unsigned lds_off, const void* gaddr,
                                                 unsigned tile_d0, unsigned tile_d1,
                                                 unsigned tensor_d0, unsigned tensor_d1,
                                                 unsigned long long stride0) {
  unsigned long long ga = (unsigned long long)(uintptr_t)gaddr;
  u32x4 g0;
  g0.x = 1u;                                                  // count=1 (valid), user mode
  g0.y = lds_off;                                             // LDS byte address
  g0.z = (unsigned)ga;                                        // global_addr[31:0]
  g0.w = (unsigned)((ga >> 32) & 0x01FFFFFFu) | 0x80000000u;  // addr[56:32] | type=2
  i32x8 g1;
  g1[0] = (int)(1u << 16);                                    // data_size = 2 bytes
  g1[1] = (int)((tensor_d0 & 0xFFFFu) << 16);                 // tensor_dim0[15:0]
  g1[2] = (int)((tensor_d0 >> 16) | ((tensor_d1 & 0xFFFFu) << 16));
  g1[3] = (int)((tensor_d1 >> 16) | ((tile_d0 & 0xFFFFu) << 16));
  g1[4] = (int)(tile_d1 & 0xFFFFu);                           // tile_dim1 (tile_dim2=0)
  g1[5] = (int)(unsigned)(stride0 & 0xFFFFFFFFu);             // dim0_stride[31:0]
  g1[6] = (int)(unsigned)((stride0 >> 32) & 0xFFFFu);         // dim0_stride[47:32]
  g1[7] = 0;
  i32x4 z4 = {0, 0, 0, 0};
  i32x8 z8 = {0, 0, 0, 0, 0, 0, 0, 0};
  __builtin_amdgcn_tensor_load_to_lds(g0, g1, z4, z4, z8, 0);
}

// ---- kernel 0: fp32 -> bf16 conversion ----------------------------------
__global__ void cvt_bf16_kernel(const float* __restrict__ in,
                                unsigned short* __restrict__ out, int n4) {
  int i = blockIdx.x * blockDim.x + threadIdx.x;
  if (i < n4) {
    float4 f = ((const float4*)in)[i];
    unsigned int lo = (unsigned int)bf16rte(f.x) | ((unsigned int)bf16rte(f.y) << 16);
    unsigned int hi = (unsigned int)bf16rte(f.z) | ((unsigned int)bf16rte(f.w) << 16);
    ((uint2*)out)[i] = make_uint2(lo, hi);
  }
}

// ---- kernel 1: projection GEMM  P = X @ W^T  (all bf16 in, bf16 out) ----
// mode 0: write [b,h,s,d] (Qp/Kp);  mode 1: write [b,h,d,s] (Vt, pre-transposed)
__global__ __launch_bounds__(256) void proj_kernel(const unsigned short* __restrict__ X,
                                                   const unsigned short* __restrict__ W,
                                                   unsigned short* __restrict__ out,
                                                   int mode) {
  const int lane = threadIdx.x & 31;
  const int wid  = threadIdx.x >> 5;
  const int Mb = blockIdx.x * 128 + (wid >> 1) * 32;
  const int Nb = blockIdx.y * 128 + (wid & 1) * 64;

  v8f acc[2][4];
#pragma unroll
  for (int i = 0; i < 2; i++)
#pragma unroll
    for (int j = 0; j < 4; j++) acc[i][j] = (v8f){0.f,0.f,0.f,0.f,0.f,0.f,0.f,0.f};

  for (int k0 = 0; k0 < DM; k0 += 32) {
    v16bf a0 = loadA_bf16(X, DM, Mb,      k0);
    v16bf a1 = loadA_bf16(X, DM, Mb + 16, k0);
#pragma unroll
    for (int j = 0; j < 4; j++) {
      v16bf b = loadB_bf16(W, DM, Nb + 16 * j, k0);
      acc[0][j] = wmma_bf16(a0, b, acc[0][j]);
      acc[1][j] = wmma_bf16(a1, b, acc[1][j]);
    }
  }

  const int halo = (lane >> 4) << 3;
#pragma unroll
  for (int i = 0; i < 2; i++)
#pragma unroll
    for (int j = 0; j < 4; j++)
#pragma unroll
      for (int r = 0; r < 8; r++) {
        int row = Mb + i * 16 + r + halo;          // b*S + s
        int col = Nb + j * 16 + (lane & 15);       // h*64 + d
        int b = row >> 11, s = row & (SEQ - 1);
        int h = col >> 6,  d = col & (DKH - 1);
        unsigned short bv = bf16rte(acc[i][j][r]);
        size_t idx;
        if (mode == 0) idx = ((size_t)(b * NH + h) * SEQ + s) * DKH + d;
        else           idx = ((size_t)(b * NH + h) * DKH + d) * SEQ + s;
        out[idx] = bv;
      }
}

// ---- kernel 2: flash attention, double-buffered TDM K/V staging ---------
// Qp,Kp: [B,H,S,64] bf16.  Vt: [B,H,64,S] bf16.  ctx: [B,S,1024] bf16.
// Block: 8 waves, one 16-query block each.  Wave0 prefetches key-block j+1
// into the ping-pong LDS buffer while all waves compute block j; TENSORcnt
// is in-order, so s_wait_tensorcnt(2) waits only for block-j's 2 transfers.
__global__ __launch_bounds__(256) void attn_kernel(const unsigned short* __restrict__ Qp,
                                                   const unsigned short* __restrict__ Kp,
                                                   const unsigned short* __restrict__ Vt,
                                                   unsigned short* __restrict__ ctx) {
  __shared__ unsigned short Ksh[2][32 * DKH];   // 2 x 4 KB: [key][dk]
  __shared__ unsigned short Vsh[2][DKH * 32];   // 2 x 4 KB: [dk][key]
  __shared__ unsigned short Pl[8][16 * 32];     // 8 KB: per-wave P staging

  const int lane = threadIdx.x & 31;
  const int wid  = threadIdx.x >> 5;
  const int bh   = blockIdx.x;
  const int qbase = blockIdx.y * 128 + wid * 16;

  const unsigned short* Qb = Qp + (size_t)bh * SEQ * DKH;
  const unsigned short* Kb = Kp + (size_t)bh * SEQ * DKH;
  const unsigned short* Vb = Vt + (size_t)bh * DKH * SEQ;

  v16bf qa0 = loadA_bf16(Qb, DKH, qbase, 0);
  v16bf qa1 = loadA_bf16(Qb, DKH, qbase, 32);

  v8f O[4];
#pragma unroll
  for (int t = 0; t < 4; t++) O[t] = (v8f){0.f,0.f,0.f,0.f,0.f,0.f,0.f,0.f};
  float m_run[8], l_run[8];
#pragma unroll
  for (int r = 0; r < 8; r++) { m_run[r] = -3.0e38f; l_run[r] = 0.f; }

  const float SCALE = 0.125f;                  // 1/sqrt(64)
  const float L2E   = 1.4426950408889634f;
  const int   NJ    = SEQ / 32;

  // prologue: wave0 fetches key-block 0 into buffer 0
  if (wid == 0) {
    tdm_load_2d_bf16((unsigned)(uintptr_t)(void*)Ksh[0], Kb, DKH, 32, DKH, SEQ, DKH);
    tdm_load_2d_bf16((unsigned)(uintptr_t)(void*)Vsh[0], Vb, 32, DKH, SEQ, DKH, SEQ);
  }

  for (int jb = 0; jb < NJ; jb++) {
    const int cur = jb & 1;
    const int nxt = cur ^ 1;
    if (wid == 0) {
      // prefetch next key-block (wraparound dummy on last iteration)
      const int jn = (jb + 1 < NJ) ? (jb + 1) : 0;
      tdm_load_2d_bf16((unsigned)(uintptr_t)(void*)Ksh[nxt],
                       Kb + (size_t)jn * 32 * DKH, DKH, 32, DKH, SEQ, DKH);
      tdm_load_2d_bf16((unsigned)(uintptr_t)(void*)Vsh[nxt],
                       Vb + (size_t)jn * 32, 32, DKH, SEQ, DKH, SEQ);
      __builtin_amdgcn_s_wait_tensorcnt(2);   // current block's 2 transfers done
    }
    __syncthreads();   // publish current K/V tiles to all waves

    // scores S[16 x 32] = Q(16x64) @ K^T from LDS
    v8f s0 = (v8f){0.f,0.f,0.f,0.f,0.f,0.f,0.f,0.f};
    v8f s1 = (v8f){0.f,0.f,0.f,0.f,0.f,0.f,0.f,0.f};
    s0 = wmma_bf16(qa0, loadB_bf16(Ksh[cur], DKH, 0,  0),  s0);
    s0 = wmma_bf16(qa1, loadB_bf16(Ksh[cur], DKH, 0,  32), s0);
    s1 = wmma_bf16(qa0, loadB_bf16(Ksh[cur], DKH, 16, 0),  s1);
    s1 = wmma_bf16(qa1, loadB_bf16(Ksh[cur], DKH, 16, 32), s1);

    float alpha[8];
#pragma unroll
    for (int r = 0; r < 8; r++) {
      float a = s0[r] * SCALE, b = s1[r] * SCALE;
      float mx = rmax16(fmaxf(a, b));
      float mn = fmaxf(m_run[r], mx);
      alpha[r] = exp2f((m_run[r] - mn) * L2E);
      float p0 = exp2f((a - mn) * L2E);
      float p1 = exp2f((b - mn) * L2E);
      s0[r] = p0; s1[r] = p1;
      float rs = rsum16(p0 + p1);
      l_run[r] = l_run[r] * alpha[r] + rs;
      m_run[r] = mn;
    }

#pragma unroll
    for (int t = 0; t < 4; t++)
#pragma unroll
      for (int r = 0; r < 8; r++) O[t][r] *= alpha[r];

    // relayout P (accumulator layout -> A layout) through LDS
    const int halo = (lane >> 4) << 3;
#pragma unroll
    for (int r = 0; r < 8; r++) {
      int m = r + halo, n = lane & 15;
      Pl[wid][m * 32 + n]      = bf16rte(s0[r]);
      Pl[wid][m * 32 + 16 + n] = bf16rte(s1[r]);
    }
    asm volatile("s_wait_dscnt 0" ::: "memory");

    ABu pa;
    {
      int m = lane & 15, h = lane >> 4;
      const unsigned short* p = &Pl[wid][m * 32 + 8 * h];
      pa.q[0] = *(const uint4*)(p);
      pa.q[1] = *(const uint4*)(p + 16);
    }

    // O += P(16x32) @ V(32x64) with V tiles from LDS
#pragma unroll
    for (int t = 0; t < 4; t++) {
      v16bf vb = loadB_bf16(Vsh[cur], 32, t * 16, 0);
      O[t] = wmma_bf16(pa.v, vb, O[t]);
    }

    __syncthreads();   // all reads of buf[cur] done before it is refilled
  }

  const int b = bh >> 4, h = bh & (NH - 1);
  const int halo = (lane >> 4) << 3;
#pragma unroll
  for (int t = 0; t < 4; t++)
#pragma unroll
    for (int r = 0; r < 8; r++) {
      int row = qbase + r + halo;
      int col = t * 16 + (lane & 15);
      float val = O[t][r] / l_run[r];
      ctx[((size_t)b * SEQ + row) * DM + h * DKH + col] = bf16rte(val);
    }
}

// ---- kernel 3: output projection  out = ctx @ Wo^T  (fp32 out) ----------
__global__ __launch_bounds__(256) void oproj_kernel(const unsigned short* __restrict__ Cx,
                                                    const unsigned short* __restrict__ W,
                                                    float* __restrict__ out) {
  const int lane = threadIdx.x & 31;
  const int wid  = threadIdx.x >> 5;
  const int Mb = blockIdx.x * 128 + (wid >> 1) * 32;
  const int Nb = blockIdx.y * 128 + (wid & 1) * 64;

  v8f acc[2][4];
#pragma unroll
  for (int i = 0; i < 2; i++)
#pragma unroll
    for (int j = 0; j < 4; j++) acc[i][j] = (v8f){0.f,0.f,0.f,0.f,0.f,0.f,0.f,0.f};

  for (int k0 = 0; k0 < DM; k0 += 32) {
    v16bf a0 = loadA_bf16(Cx, DM, Mb,      k0);
    v16bf a1 = loadA_bf16(Cx, DM, Mb + 16, k0);
#pragma unroll
    for (int j = 0; j < 4; j++) {
      v16bf b = loadB_bf16(W, DM, Nb + 16 * j, k0);
      acc[0][j] = wmma_bf16(a0, b, acc[0][j]);
      acc[1][j] = wmma_bf16(a1, b, acc[1][j]);
    }
  }

  const int halo = (lane >> 4) << 3;
#pragma unroll
  for (int i = 0; i < 2; i++)
#pragma unroll
    for (int j = 0; j < 4; j++)
#pragma unroll
      for (int r = 0; r < 8; r++) {
        int row = Mb + i * 16 + r + halo;
        int col = Nb + j * 16 + (lane & 15);
        out[(size_t)row * DM + col] = acc[i][j][r];
      }
}

// ---- host orchestration -------------------------------------------------
extern "C" void kernel_launch(void* const* d_in, const int* in_sizes, int n_in,
                              void* d_out, int out_size, void* d_ws, size_t ws_size,
                              hipStream_t stream) {
  (void)in_sizes; (void)n_in; (void)out_size; (void)ws_size;
  const float* q  = (const float*)d_in[0];
  const float* k  = (const float*)d_in[1];
  const float* v  = (const float*)d_in[2];
  const float* wq = (const float*)d_in[3];
  const float* wk = (const float*)d_in[4];
  const float* wv = (const float*)d_in[5];
  const float* wo = (const float*)d_in[6];

  char* ws = (char*)d_ws;
  const size_t MB = 1u << 20;
  unsigned short* wqb = (unsigned short*)(ws +  0 * MB);   // 2 MB each
  unsigned short* wkb = (unsigned short*)(ws +  2 * MB);
  unsigned short* wvb = (unsigned short*)(ws +  4 * MB);
  unsigned short* wob = (unsigned short*)(ws +  6 * MB);
  unsigned short* qb  = (unsigned short*)(ws +  8 * MB);   // 16 MB each
  unsigned short* kb  = (unsigned short*)(ws + 24 * MB);
  unsigned short* vb  = (unsigned short*)(ws + 40 * MB);
  unsigned short* Qp  = (unsigned short*)(ws + 56 * MB);
  unsigned short* Kp  = (unsigned short*)(ws + 72 * MB);
  unsigned short* Vt  = (unsigned short*)(ws + 88 * MB);
  unsigned short* Cx  = (unsigned short*)(ws + 104 * MB);

  dim3 cb(256);
  {
    const int n4 = (DM * DM) / 4;
    dim3 cg((n4 + 255) / 256);
    cvt_bf16_kernel<<<cg, cb, 0, stream>>>(wq, wqb, n4);
    cvt_bf16_kernel<<<cg, cb, 0, stream>>>(wk, wkb, n4);
    cvt_bf16_kernel<<<cg, cb, 0, stream>>>(wv, wvb, n4);
    cvt_bf16_kernel<<<cg, cb, 0, stream>>>(wo, wob, n4);
  }
  {
    const int n4 = (NB * SEQ * DM) / 4;
    dim3 cg((n4 + 255) / 256);
    cvt_bf16_kernel<<<cg, cb, 0, stream>>>(q, qb, n4);
    cvt_bf16_kernel<<<cg, cb, 0, stream>>>(k, kb, n4);
    cvt_bf16_kernel<<<cg, cb, 0, stream>>>(v, vb, n4);
  }

  dim3 gg(NB * SEQ / 128, DM / 128), gb(256);
  proj_kernel<<<gg, gb, 0, stream>>>(qb, wqb, Qp, 0);
  proj_kernel<<<gg, gb, 0, stream>>>(kb, wkb, Kp, 0);
  proj_kernel<<<gg, gb, 0, stream>>>(vb, wvb, Vt, 1);

  dim3 ag(NB * NH, SEQ / 128);
  attn_kernel<<<ag, gb, 0, stream>>>(Qp, Kp, Vt, Cx);

  oproj_kernel<<<gg, gb, 0, stream>>>(Cx, wob, (float*)d_out);
}
